// NeuralNetwork_84104049590322
// MI455X (gfx1250) — matrix operand order repaired and checked by reading, not compiled, not run on hardware
//
#include <hip/hip_runtime.h>
#include <hip/hip_bf16.h>

// ---------------------------------------------------------------------------
// Deep-hedging MLP rollout for MI455X (gfx1250).
//
// Strategy (compute-bound: ~1.3 TFLOP vs ~30MB HBM traffic):
//  * f16 WMMA (v_wmma_f32_16x16x32_f16), f32 accumulate.
//  * All MLP weights converted to f16 and pre-swizzled into WMMA B-fragment
//    tile order in the workspace by a prep kernel, then staged once into LDS
//    per workgroup (~204KB; fits in the 320KB WGP LDS).
//  * Persistent workgroup per 128-path tile, 256 threads (8 wave32),
//    wave w owns M-rows [16w,16w+16). Activations ping-pong in LDS in
//    A-fragment layout; matmul epilogue fuses bias+ReLU+cvt+re-layout.
// ---------------------------------------------------------------------------

typedef _Float16 half_t;
typedef __attribute__((ext_vector_type(16))) _Float16 v16h;
typedef __attribute__((ext_vector_type(8)))  float    v8f;

#define NBATCH   65536
#define NSTEPS   100
#define FSTEP    0.01f
#define WID      128
#define M_TILE   128   // batch rows per workgroup
#define NTHREADS 256   // 8 wave32

// Fragment-tile sizes: each 32(K)x16(N) weight tile = 32 lanes * 16 halves.
#define TILE_HALVES 512

// Workspace layout (in halves) for the swizzled f16 weights.
#define OFF_W1_0 0                         // (4->pad32) x 128 : 1 kt x 8 nt
#define OFF_W2_0 (OFF_W1_0 + 1*8*TILE_HALVES)
#define OFF_W1_1 (OFF_W2_0 + 4*8*TILE_HALVES)
#define OFF_W2_1 (OFF_W1_1 + 4*8*TILE_HALVES)
#define OFF_W1_2 (OFF_W2_1 + 4*8*TILE_HALVES)
#define OFF_W2_2 (OFF_W1_2 + 4*8*TILE_HALVES)
#define OFF_W1_3 (OFF_W2_2 + 4*8*TILE_HALVES)
#define OFF_W2_3 (OFF_W1_3 + 4*8*TILE_HALVES)  // 128 x (2->pad16): 4 kt x 1 nt
#define W_TOTAL  (OFF_W2_3 + 4*1*TILE_HALVES)  // = 104448 halves = 208896 B

// Dynamic LDS layout (bytes).
#define SM_W_BYTES     (W_TOTAL * 2)            // 208896
#define SM_BIAS_BYTES  (8 * 128 * 4)            // 4096  (8 bias rows, f32)
#define SM_IN_BYTES    (8 * TILE_HALVES * 2)    // 8192  (nn_in A-frag, K=32 pad)
#define SM_ACT_BYTES   (32 * TILE_HALVES * 2)   // 32768 (128x128 A-frag)
#define SM_OUT2_BYTES  (M_TILE * 2 * 4)         // 1024
#define SMEM_BYTES (SM_W_BYTES + SM_BIAS_BYTES + SM_IN_BYTES + 2*SM_ACT_BYTES + SM_OUT2_BYTES) // 287744

// ---------------------------------------------------------------------------
// Prep: convert f32 weight (realK x realN, row-major) to f16 tiles in WMMA
// B-fragment order. Tile (kt,nt), lane l, half h holds element
//   K = 32*kt + 8*(l>>4) + (h&7) + 16*(h>>3) ,  N = 16*nt + (l&15)
// stored at dst[((kt*NT + nt)*32 + l)*16 + h]. Out-of-range -> 0 (padding).
// ---------------------------------------------------------------------------
__global__ void prep_weights(const float* __restrict__ src, half_t* __restrict__ dst,
                             int realK, int realN, int KT, int NT) {
  int total = KT * NT * TILE_HALVES;
  for (int i = blockIdx.x * blockDim.x + threadIdx.x; i < total;
       i += gridDim.x * blockDim.x) {
    int h = i & 15;
    int l = (i >> 4) & 31;
    int tile = i >> 9;
    int nt = tile % NT;
    int kt = tile / NT;
    int k = 32 * kt + 8 * (l >> 4) + (h & 7) + 16 * (h >> 3);
    int n = 16 * nt + (l & 15);
    float v = (k < realK && n < realN) ? src[k * realN + n] : 0.0f;
    dst[i] = (half_t)v;
  }
}

// ---------------------------------------------------------------------------
// One dense layer on the 128-row tile.
//  src : A-fragment activations, layout [mt][KT][lane][16] (mt = wave)
//  wt  : B-fragment weights,     layout [kt][NT][lane][16]
//  dst : next A-fragment buffer (KT'=4) -- epilogue scatters bias+ReLU+f16.
//  OUT2: last layer writes f32 (leverage,hedge) pairs instead.
// A-fragment half->K map (16-bit A 16x32, ISA 7.12.2):
//   K = ((j<4)?2*j:16+2*(j-4)) + 8*(lane>>4) + (h&1), j = h>>1.
// ---------------------------------------------------------------------------
template <int KT, int NT, bool RELU, bool OUT2>
__device__ inline void mlp_layer(const half_t* __restrict__ src,
                                 const half_t* __restrict__ wt,
                                 const float* __restrict__ bias,
                                 half_t* __restrict__ dst,
                                 float* __restrict__ out2,
                                 int wave, int lane) {
  v16h a[KT];
#pragma unroll
  for (int kt = 0; kt < KT; ++kt)
    a[kt] = *(const v16h*)(src + (((wave * KT + kt) * 32 + lane) << 4));

  const int n   = lane & 15;
  const int g16 = lane >> 4;

#pragma unroll
  for (int nt = 0; nt < NT; ++nt) {
    v8f c = {};
#pragma unroll
    for (int kt = 0; kt < KT; ++kt) {
      v16h b = *(const v16h*)(wt + (((kt * NT + nt) * 32 + lane) << 4));
      c = __builtin_amdgcn_wmma_f32_16x16x32_f16(
          /*neg_a=*/false, a[kt], /*neg_b=*/false, b,
          /*c_mod=*/(short)0, c, /*reuse_a=*/false, /*reuse_b=*/false);
    }
    const int gcol = nt * 16 + n;     // output column (== next layer K index)
    const float bv = bias[gcol];
    if (OUT2) {
      if (n < 2) {
#pragma unroll
        for (int v = 0; v < 8; ++v) {
          int row = wave * 16 + v + 8 * g16;   // C layout: M = vgpr + 8*(lane>=16)
          out2[row * 2 + n] = c[v] + bv;
        }
      }
    } else {
      // Scatter into destination A-fragment layout (dest KT' = 4).
      const int ktd = gcol >> 5;
      const int kk  = gcol & 31;
      const int g   = (kk >> 3) & 1;                     // lane-group of dest
      const int hi  = kk >> 4;
      const int r   = kk & 7;
      const int h   = 2 * (4 * hi + (r >> 1)) + (kk & 1);
      const int base = ((wave * 4 + ktd) * 32) * 16 + h; // + destLane*16
#pragma unroll
      for (int v = 0; v < 8; ++v) {
        float x = c[v] + bv;
        if (RELU) x = fmaxf(x, 0.0f);
        int dl = (v + 8 * g16) + 16 * g;                 // dest lane
        dst[base + dl * 16] = (half_t)x;
      }
    }
  }
}

// ---------------------------------------------------------------------------
// Main persistent kernel: one WG = 128 paths, all 100 steps.
// ---------------------------------------------------------------------------
__global__ void __launch_bounds__(NTHREADS)
deep_hedge(const float* __restrict__ S0, const float* __restrict__ Kp,
           const float* __restrict__ Tp, const float* __restrict__ BM,
           const half_t* __restrict__ wfrag,
           const float* __restrict__ b10, const float* __restrict__ b20,
           const float* __restrict__ b11, const float* __restrict__ b21,
           const float* __restrict__ b12, const float* __restrict__ b22,
           const float* __restrict__ b13, const float* __restrict__ b23,
           float* __restrict__ out) {
  extern __shared__ char smem[];
  half_t* sW    = (half_t*)smem;
  float*  sBias = (float*)(smem + SM_W_BYTES);
  half_t* sIn   = (half_t*)(smem + SM_W_BYTES + SM_BIAS_BYTES);
  half_t* sAct0 = sIn + 8 * TILE_HALVES;
  half_t* sAct1 = sAct0 + 32 * TILE_HALVES;
  float*  sOut2 = (float*)(sAct1 + 32 * TILE_HALVES);

  const int tid  = threadIdx.x;
  const int wave = tid >> 5;
  const int lane = tid & 31;

  // ---- stage all swizzled weights into LDS (one-time, 16B vector copies) ---
  {
    const int4* wsrc = (const int4*)wfrag;
    int4* wdst = (int4*)sW;
    for (int i = tid; i < W_TOTAL / 8; i += NTHREADS) wdst[i] = wsrc[i];
  }
  // ---- biases (f32), last layer padded to 128 with zeros ----
  for (int i = tid; i < 128; i += NTHREADS) {
    sBias[0 * 128 + i] = b10[i];
    sBias[1 * 128 + i] = b20[i];
    sBias[2 * 128 + i] = b11[i];
    sBias[3 * 128 + i] = b21[i];
    sBias[4 * 128 + i] = b12[i];
    sBias[5 * 128 + i] = b22[i];
    sBias[6 * 128 + i] = b13[i];
    sBias[7 * 128 + i] = (i < 2) ? b23[i] : 0.0f;
  }
  // ---- zero nn_in fragment buffer once (K padding 4..31 must be 0, not NaN)
  for (int i = tid; i < 8 * TILE_HALVES; i += NTHREADS) sIn[i] = (half_t)0.0f;
  __syncthreads();

  // ---- per-path state (threads 0..127 own one path each) ----
  const int rowG = blockIdx.x * M_TILE + tid;
  float price = 1.0f, strike = 1.0f, mat = 0.0f, hedgepf = 0.0f;
  if (tid < M_TILE) {
    price  = S0[rowG];
    strike = Kp[rowG];
    mat    = Tp[rowG];
  }

#pragma unroll 1
  for (int step = 0; step < NSTEPS; ++step) {
    const float t = (float)step * (1.0f / NSTEPS);

    // nn_in = [price, t, T-t, log(price/K)], written as A-fragment K=0..3
    // (lane = row&15, halves h=0..3 hold K=h for lane-group 0).
    if (tid < M_TILE) {
      const int base = (((tid >> 4) * 32) + (tid & 15)) * 16;
      sIn[base + 0] = (half_t)price;
      sIn[base + 1] = (half_t)t;
      sIn[base + 2] = (half_t)(mat - t);
      sIn[base + 3] = (half_t)__logf(price / strike);
      if (step + 1 < NSTEPS)  // gfx1250 global_prefetch of next BM increment
        __builtin_prefetch(&BM[rowG * NSTEPS + step + 1], 0, 0);
    }
    __syncthreads();

    // ---- 4-stack MLP, 204 v_wmma per wave per step ----
    mlp_layer<1, 8, true,  false>(sIn,   sW + OFF_W1_0, sBias + 0 * 128, sAct0, nullptr, wave, lane); __syncthreads();
    mlp_layer<4, 8, false, false>(sAct0, sW + OFF_W2_0, sBias + 1 * 128, sAct1, nullptr, wave, lane); __syncthreads();
    mlp_layer<4, 8, true,  false>(sAct1, sW + OFF_W1_1, sBias + 2 * 128, sAct0, nullptr, wave, lane); __syncthreads();
    mlp_layer<4, 8, false, false>(sAct0, sW + OFF_W2_1, sBias + 3 * 128, sAct1, nullptr, wave, lane); __syncthreads();
    mlp_layer<4, 8, true,  false>(sAct1, sW + OFF_W1_2, sBias + 4 * 128, sAct0, nullptr, wave, lane); __syncthreads();
    mlp_layer<4, 8, false, false>(sAct0, sW + OFF_W2_2, sBias + 5 * 128, sAct1, nullptr, wave, lane); __syncthreads();
    mlp_layer<4, 8, true,  false>(sAct1, sW + OFF_W1_3, sBias + 6 * 128, sAct0, nullptr, wave, lane); __syncthreads();
    mlp_layer<4, 1, false, true >(sAct0, sW + OFF_W2_3, sBias + 7 * 128, nullptr, sOut2, wave, lane); __syncthreads();

    // ---- SDE / hedge update ----
    if (tid < M_TILE) {
      const float lev = sOut2[tid * 2 + 0];
      const float hed = sOut2[tid * 2 + 1];
      const float bm  = BM[rowG * NSTEPS + step];
      const float docomp = (t <= mat) ? 1.0f : 0.0f;
      const float dS = lev * price * (bm * FSTEP);
      price   += docomp * dS;
      hedgepf += docomp * (hed * dS);
    }
    // sOut2 reuse is safe: next write is 8 barriers away.
  }

  if (tid < M_TILE) {
    const float payoff = fmaxf(price - strike, 0.0f);  // (x+|x|)/2 == relu
    out[rowG] = payoff - hedgepf;
  }
}

// ---------------------------------------------------------------------------
// Host launcher.
// Input order (setup_inputs dict, params tuple flattened):
//  0:S0 1:K 2:T 3:BM  then per stack s: 4+4s:W1  5+4s:b1  6+4s:W2  7+4s:b2
// ---------------------------------------------------------------------------
extern "C" void kernel_launch(void* const* d_in, const int* in_sizes, int n_in,
                              void* d_out, int out_size, void* d_ws, size_t ws_size,
                              hipStream_t stream) {
  const float* S0 = (const float*)d_in[0];
  const float* Kp = (const float*)d_in[1];
  const float* Tp = (const float*)d_in[2];
  const float* BM = (const float*)d_in[3];
  half_t* wfrag = (half_t*)d_ws;  // needs 208896 B of workspace

  // Swizzle + f16-convert all weights (deterministic, cheap, every call).
  prep_weights<<<64, 256, 0, stream>>>((const float*)d_in[4],  wfrag + OFF_W1_0,   4, 128, 1, 8);
  prep_weights<<<64, 256, 0, stream>>>((const float*)d_in[6],  wfrag + OFF_W2_0, 128, 128, 4, 8);
  prep_weights<<<64, 256, 0, stream>>>((const float*)d_in[8],  wfrag + OFF_W1_1, 128, 128, 4, 8);
  prep_weights<<<64, 256, 0, stream>>>((const float*)d_in[10], wfrag + OFF_W2_1, 128, 128, 4, 8);
  prep_weights<<<64, 256, 0, stream>>>((const float*)d_in[12], wfrag + OFF_W1_2, 128, 128, 4, 8);
  prep_weights<<<64, 256, 0, stream>>>((const float*)d_in[14], wfrag + OFF_W2_2, 128, 128, 4, 8);
  prep_weights<<<64, 256, 0, stream>>>((const float*)d_in[16], wfrag + OFF_W1_3, 128, 128, 4, 8);
  prep_weights<<<64, 256, 0, stream>>>((const float*)d_in[18], wfrag + OFF_W2_3, 128,   2, 4, 1);

  // Allow >64KB dynamic LDS (287744 B <= 320KB WGP LDS). Idempotent.
  (void)hipFuncSetAttribute(reinterpret_cast<const void*>(&deep_hedge),
                            hipFuncAttributeMaxDynamicSharedMemorySize,
                            (int)SMEM_BYTES);

  deep_hedge<<<NBATCH / M_TILE, NTHREADS, SMEM_BYTES, stream>>>(
      S0, Kp, Tp, BM, wfrag,
      (const float*)d_in[5],  (const float*)d_in[7],
      (const float*)d_in[9],  (const float*)d_in[11],
      (const float*)d_in[13], (const float*)d_in[15],
      (const float*)d_in[17], (const float*)d_in[19],
      (float*)d_out);
}
